// LSTMDSTGCRN_21509196219017
// MI455X (gfx1250) — matrix operand order repaired and checked by reading, not compiled, not run on hardware
//
#include <hip/hip_runtime.h>

// ---------------------------------------------------------------------------
// LSTM-DSTGCRN for MI455X (gfx1250, wave32, WMMA).
//
// Dominant cost: 16 sequential GEMMs  AHX[2048,1088] = A[2048,2048] @ HX[2048,1088]
// (146 GFLOP total). A (bf16, 8MB), H-transposed (bf16, 4MB), packed X (4MB)
// all live in the 192MB L2, so the GEMM is matrix-core bound -> use
// v_wmma_f32_16x16x32_bf16 with f32 accumulation, double-buffered LDS tiles
// filled via GLOBAL_LOAD_ASYNC_TO_LDS_B128 (ASYNCcnt) when available.
// ---------------------------------------------------------------------------

#define NNODES  2048
#define CIN     2
#define HID     32
#define EMBD    16
#define HORIZON 12
#define BATCH   32
#define TSTEPS  16
#define BH      (BATCH * HID)   // 1024  h-columns
#define BC      (BATCH * CIN)   // 64    x-columns
#define NCOLS   (BH + BC)       // 1088  GEMM column count
#define KTILES  (NNODES / 32)   // 64 K steps

typedef __attribute__((ext_vector_type(16))) __bf16       v16bf;
typedef __attribute__((ext_vector_type(8)))  float        v8f;
typedef __attribute__((ext_vector_type(4)))  unsigned int u32x4;
typedef __attribute__((ext_vector_type(4)))  int          v4i;

union Frag16 { v16bf v; u32x4 q[2]; };

#ifndef __has_builtin
#define __has_builtin(x) 0
#endif
#if __has_builtin(__builtin_amdgcn_global_load_async_to_lds_b128)
#define HAVE_ASYNC_LDS 1
#else
#define HAVE_ASYNC_LDS 0
#endif

#if HAVE_ASYNC_LDS
#if __has_builtin(__builtin_amdgcn_s_wait_asynccnt)
#define WAIT_ASYNC0() __builtin_amdgcn_s_wait_asynccnt(0)
#else
#define WAIT_ASYNC0() asm volatile("s_wait_asynccnt 0x0" ::: "memory")
#endif
// Builtin signature (from clang diagnostic): src is AS(1) pointer to int4
// vector; dst is the LDS (AS(3)) counterpart. Flat LDS addresses carry the
// LDS byte offset in bits [31:0] (aperture rules), so integer truncation
// yields a valid AS(3) pointer.
typedef __attribute__((address_space(1))) v4i* gv4i_p;
typedef __attribute__((address_space(3))) v4i* lv4i_p;
// Copy 32 contiguous bytes global -> LDS via two async b128 transfers.
__device__ __forceinline__ void async_cp32B(const unsigned short* g,
                                            unsigned short* l) {
  gv4i_p gp = (gv4i_p)(unsigned long long)g;
  lv4i_p lp = (lv4i_p)(unsigned int)(unsigned long long)l;
  __builtin_amdgcn_global_load_async_to_lds_b128(gp, lp, 0, 0);
  __builtin_amdgcn_global_load_async_to_lds_b128(gp, lp, 16, 0);
}
#endif

__device__ __forceinline__ unsigned short f32_to_bf16(float f) {
  union { float f; unsigned int u; } v; v.f = f;
  unsigned int r = v.u + 0x7FFFu + ((v.u >> 16) & 1u);   // round-to-nearest-even
  return (unsigned short)(r >> 16);
}
__device__ __forceinline__ float bf16_to_f32(unsigned short h) {
  union { unsigned int u; float f; } v; v.u = ((unsigned int)h) << 16;
  return v.f;
}

// ---------------------------------------------------------------------------
__global__ void zero_u32_kernel(unsigned int* __restrict__ p, int n) {
  int i = blockIdx.x * blockDim.x + threadIdx.x;
  if (i < n) p[i] = 0u;
}

// ---------------------------------------------------------------------------
// A = softmax(relu(E1 @ E2^T)) row-wise, stored bf16 row-major [N][N].
// One block (256 threads) per row; each thread owns 8 columns.
__global__ __launch_bounds__(256) void adj_softmax_kernel(
    const float* __restrict__ E1, const float* __restrict__ E2,
    unsigned short* __restrict__ Abf) {
  const int n = blockIdx.x, tid = threadIdx.x;
  __shared__ float e1[EMBD];
  __shared__ float red[256];
  if (tid < EMBD) e1[tid] = E1[n * EMBD + tid];
  __syncthreads();

  float s[8];
  float m = 0.f;                       // relu >= 0 so max >= 0
  #pragma unroll
  for (int it = 0; it < 8; ++it) {
    const int j = it * 256 + tid;
    const float* e2 = E2 + (size_t)j * EMBD;
    float d = 0.f;
    #pragma unroll
    for (int q = 0; q < EMBD; ++q) d = fmaf(e1[q], e2[q], d);
    d = d > 0.f ? d : 0.f;
    s[it] = d;
    m = fmaxf(m, d);
  }
  red[tid] = m; __syncthreads();
  for (int off = 128; off > 0; off >>= 1) {
    if (tid < off) red[tid] = fmaxf(red[tid], red[tid + off]);
    __syncthreads();
  }
  m = red[0]; __syncthreads();

  float ls = 0.f;
  #pragma unroll
  for (int it = 0; it < 8; ++it) { s[it] = __expf(s[it] - m); ls += s[it]; }
  red[tid] = ls; __syncthreads();
  for (int off = 128; off > 0; off >>= 1) {
    if (tid < off) red[tid] += red[tid + off];
    __syncthreads();
  }
  const float inv = 1.f / red[0];
  #pragma unroll
  for (int it = 0; it < 8; ++it)
    Abf[(size_t)n * NNODES + it * 256 + tid] = f32_to_bf16(s[it] * inv);
}

// ---------------------------------------------------------------------------
// Pack x [B,T,N,C] f32 -> XallT [T][B*C][N] bf16 (transposed: contiguous K for
// WMMA B-fragments).
__global__ __launch_bounds__(256) void pack_x_kernel(
    const float* __restrict__ x, unsigned short* __restrict__ XT) {
  const int row = blockIdx.y;                // t*64 + b*2 + c
  const int n   = blockIdx.x * 256 + threadIdx.x;
  const int t = row >> 6, r = row & 63, b = r >> 1, c = r & 1;
  const float v = x[(((size_t)b * TSTEPS + t) * NNODES + n) * CIN + c];
  XT[(size_t)row * NNODES + n] = f32_to_bf16(v);
}

// ---------------------------------------------------------------------------
// AHX[2048,1088] = A[2048,2048] @ [H | X_t][2048,1088]
// B operand stored transposed ([col][K]) so B-fragments are contiguous in K.
// Block: 256 threads / 8 waves; 128(M) x 64(cols) tile; wave = 32x32 microtile
// = 2x2 v_wmma_f32_16x16x32_bf16 per 32-wide K step.
// Double-buffered LDS: one workgroup barrier per K step; next tile streams in
// (async-to-LDS when available) while the current tile feeds the WMMAs.
__global__ __launch_bounds__(256) void gemm_wmma_kernel(
    const unsigned short* __restrict__ Abf,
    const unsigned short* __restrict__ HXT,   // [1024][2048] bf16 (h transposed)
    const unsigned short* __restrict__ XTt,   // [64][2048]  bf16 (x_t transposed)
    float* __restrict__ AHX) {
  __shared__ alignas(16) unsigned short Asm[2][128][40];  // +8 pad per row
  __shared__ alignas(16) unsigned short Bsm[2][64][40];

  const int tid  = threadIdx.x;
  const int m0   = blockIdx.y * 128;
  const int jc0  = blockIdx.x * 64;
  const int lane = tid & 31, w = tid >> 5;
  const int wm = w & 3, wn = w >> 2;          // wave grid 4(M) x 2(cols)
  const int lr = lane & 15, hf = lane >> 4;

  // cooperative-load roles: A tile 128x32 -> 256 thr x 32B, B 64x32 -> 128 thr
  const int arow = tid >> 1, ahalf = tid & 1;
  const unsigned short* aG = Abf + (size_t)(m0 + arow) * NNODES + ahalf * 16;
  const int bj = jc0 + (tid >> 1);            // global output column
  const unsigned short* bRow =
      (bj < BH) ? (HXT + (size_t)bj * NNODES)
                : (XTt + (size_t)(bj - BH) * NNODES);
  const unsigned short* bG = bRow + ahalf * 16;

  auto load_tiles = [&](int buf, int k0) {
#if HAVE_ASYNC_LDS
    async_cp32B(aG + k0, &Asm[buf][arow][ahalf * 16]);
    if (tid < 128) async_cp32B(bG + k0, &Bsm[buf][tid >> 1][ahalf * 16]);
#else
    {
      const u32x4* s = (const u32x4*)(aG + k0);
      u32x4 p0 = s[0], p1 = s[1];
      *(u32x4*)&Asm[buf][arow][ahalf * 16]     = p0;
      *(u32x4*)&Asm[buf][arow][ahalf * 16 + 8] = p1;
      __builtin_prefetch(aG + k0 + 32, 0, 1);   // global_prefetch_b8
    }
    if (tid < 128) {
      const u32x4* s = (const u32x4*)(bG + k0);
      u32x4 p0 = s[0], p1 = s[1];
      *(u32x4*)&Bsm[buf][tid >> 1][ahalf * 16]     = p0;
      *(u32x4*)&Bsm[buf][tid >> 1][ahalf * 16 + 8] = p1;
      __builtin_prefetch(bG + k0 + 32, 0, 1);
    }
#endif
  };

  v8f acc00 = {0.f,0.f,0.f,0.f,0.f,0.f,0.f,0.f};
  v8f acc01 = {0.f,0.f,0.f,0.f,0.f,0.f,0.f,0.f};
  v8f acc10 = {0.f,0.f,0.f,0.f,0.f,0.f,0.f,0.f};
  v8f acc11 = {0.f,0.f,0.f,0.f,0.f,0.f,0.f,0.f};

  load_tiles(0, 0);                       // prologue fill of buffer 0

  for (int kt = 0; kt < KTILES; ++kt) {
    const int cur = kt & 1;
#if HAVE_ASYNC_LDS
    WAIT_ASYNC0();                        // own async fills of buf[cur] landed
#endif
    __syncthreads();                      // all waves' fills of buf[cur] visible
    if (kt + 1 < KTILES) load_tiles(cur ^ 1, (kt + 1) * 32);

    // bf16 A-frag lane layout (ISA 7.12.2): row = lane%16,
    //   elems 0..7  -> K = 8*half + e, elems 8..15 -> K = 16 + 8*half + e
    // bf16 B-frag: col = lane%16, same K order as A-frag
    Frag16 a0, a1, b0, b1;
    a0.q[0] = *(const u32x4*)&Asm[cur][wm * 32 + lr][8 * hf];
    a0.q[1] = *(const u32x4*)&Asm[cur][wm * 32 + lr][16 + 8 * hf];
    a1.q[0] = *(const u32x4*)&Asm[cur][wm * 32 + 16 + lr][8 * hf];
    a1.q[1] = *(const u32x4*)&Asm[cur][wm * 32 + 16 + lr][16 + 8 * hf];
    b0.q[0] = *(const u32x4*)&Bsm[cur][wn * 32 + lr][8 * hf];
    b0.q[1] = *(const u32x4*)&Bsm[cur][wn * 32 + lr][16 + 8 * hf];
    b1.q[0] = *(const u32x4*)&Bsm[cur][wn * 32 + 16 + lr][8 * hf];
    b1.q[1] = *(const u32x4*)&Bsm[cur][wn * 32 + 16 + lr][16 + 8 * hf];

    acc00 = __builtin_amdgcn_wmma_f32_16x16x32_bf16(false, a0.v, false, b0.v,
                                                    (short)0, acc00, false, false);
    acc01 = __builtin_amdgcn_wmma_f32_16x16x32_bf16(false, a0.v, false, b1.v,
                                                    (short)0, acc01, false, false);
    acc10 = __builtin_amdgcn_wmma_f32_16x16x32_bf16(false, a1.v, false, b0.v,
                                                    (short)0, acc10, false, false);
    acc11 = __builtin_amdgcn_wmma_f32_16x16x32_bf16(false, a1.v, false, b1.v,
                                                    (short)0, acc11, false, false);
  }

  // C/D layout: N = lane%16, M = vgpr + 8*(lane/16)
  const int colb = jc0 + wn * 32 + lr;
  const int rowb = m0 + wm * 32 + hf * 8;
  #pragma unroll
  for (int v = 0; v < 8; ++v) {
    AHX[(size_t)(rowb + v) * NCOLS + colb]           = acc00[v];
    AHX[(size_t)(rowb + v) * NCOLS + colb + 16]      = acc01[v];
    AHX[(size_t)(rowb + 16 + v) * NCOLS + colb]      = acc10[v];
    AHX[(size_t)(rowb + 16 + v) * NCOLS + colb + 16] = acc11[v];
  }
}

// ---------------------------------------------------------------------------
// Gates + LSTM pointwise. 256 threads = 8 (b,n) pairs x 32 hidden lanes.
// gates[j] = sum_k AH[n][b*32+k]*Wh[k][j] + sum_c AX[n][b*2+c]*Wx[c][j] + bx+bh
__global__ __launch_bounds__(256) void gates_lstm_kernel(
    const float* __restrict__ AHX,
    const float* __restrict__ Wx, const float* __restrict__ bx,
    const float* __restrict__ Wh, const float* __restrict__ bh,
    float* __restrict__ c_state, unsigned short* __restrict__ HXT) {
  __shared__ float Whs[HID][128];
  __shared__ float Wxs[CIN][128];
  __shared__ float bsum[128];
  __shared__ float segs[8][HID];
  __shared__ float axs[8][CIN];
  const int tid = threadIdx.x;
  for (int i = tid; i < HID * 128; i += 256) (&Whs[0][0])[i] = Wh[i];
  for (int i = tid; i < CIN * 128; i += 256) (&Wxs[0][0])[i] = Wx[i];
  for (int i = tid; i < 128; i += 256) bsum[i] = bx[i] + bh[i];

  const int p = tid >> 5, lane = tid & 31;
  const int gp = blockIdx.x * 8 + p;
  const int b = gp >> 11, n = gp & (NNODES - 1);
  segs[p][lane] = AHX[(size_t)n * NCOLS + b * HID + lane];
  if (lane < CIN) axs[p][lane] = AHX[(size_t)n * NCOLS + BH + b * CIN + lane];
  __syncthreads();

  const int hh = lane;
  float g0 = bsum[hh], g1 = bsum[32 + hh], g2 = bsum[64 + hh], g3 = bsum[96 + hh];
  #pragma unroll
  for (int k = 0; k < HID; ++k) {
    const float a = segs[p][k];
    g0 = fmaf(a, Whs[k][hh],      g0);
    g1 = fmaf(a, Whs[k][32 + hh], g1);
    g2 = fmaf(a, Whs[k][64 + hh], g2);
    g3 = fmaf(a, Whs[k][96 + hh], g3);
  }
  #pragma unroll
  for (int c = 0; c < CIN; ++c) {
    const float a = axs[p][c];
    g0 = fmaf(a, Wxs[c][hh],      g0);
    g1 = fmaf(a, Wxs[c][32 + hh], g1);
    g2 = fmaf(a, Wxs[c][64 + hh], g2);
    g3 = fmaf(a, Wxs[c][96 + hh], g3);
  }
  const float ii = 1.f / (1.f + __expf(-g0));
  const float ff = 1.f / (1.f + __expf(-g1));
  const float oo = 1.f / (1.f + __expf(-g2));
  const float gg = tanhf(g3);
  const size_t ci = ((size_t)b * NNODES + n) * HID + hh;
  const float cn = ff * c_state[ci] + ii * gg;
  c_state[ci] = cn;
  const float hn = oo * tanhf(cn);
  HXT[(size_t)(b * HID + hh) * NNODES + n] = f32_to_bf16(hn);  // transposed bf16
}

// ---------------------------------------------------------------------------
// out[b, hz, n, 0] = h_last[b,n,:] @ Wp[:,hz] + bp[hz]
__global__ __launch_bounds__(256) void proj_kernel(
    const unsigned short* __restrict__ HXT,
    const float* __restrict__ Wp, const float* __restrict__ bp,
    float* __restrict__ out) {
  __shared__ float Wps[HID * HORIZON];
  __shared__ float bps[HORIZON];
  const int tid = threadIdx.x;
  for (int i = tid; i < HID * HORIZON; i += 256) Wps[i] = Wp[i];
  if (tid < HORIZON) bps[tid] = bp[tid];
  __syncthreads();
  const int b = blockIdx.y;
  const int n = blockIdx.x * 256 + tid;
  float h[HID];
  #pragma unroll
  for (int k = 0; k < HID; ++k)
    h[k] = bf16_to_f32(HXT[(size_t)(b * HID + k) * NNODES + n]);  // coalesced per k
  #pragma unroll
  for (int hz = 0; hz < HORIZON; ++hz) {
    float o = bps[hz];
    #pragma unroll
    for (int k = 0; k < HID; ++k) o = fmaf(h[k], Wps[k * HORIZON + hz], o);
    out[((size_t)b * HORIZON + hz) * NNODES + n] = o;
  }
}

// ---------------------------------------------------------------------------
extern "C" void kernel_launch(void* const* d_in, const int* in_sizes, int n_in,
                              void* d_out, int out_size, void* d_ws, size_t ws_size,
                              hipStream_t stream) {
  const float* x  = (const float*)d_in[0];
  const float* E1 = (const float*)d_in[1];
  const float* E2 = (const float*)d_in[2];
  const float* Wx = (const float*)d_in[3];
  const float* bx = (const float*)d_in[4];
  const float* Wh = (const float*)d_in[5];
  const float* bh = (const float*)d_in[6];
  const float* Wp = (const float*)d_in[7];
  const float* bp = (const float*)d_in[8];
  float* out = (float*)d_out;

  // workspace layout (256B aligned offsets), ~33.5 MB total
  char* ws = (char*)d_ws;
  unsigned short* Abf   = (unsigned short*)(ws + 0);          // 2048*2048 bf16 = 8 MB
  unsigned short* XallT = (unsigned short*)(ws + 8388608);    // 16*64*2048 bf16 = 4 MB
  unsigned short* HXT   = (unsigned short*)(ws + 12582912);   // 1024*2048 bf16 = 4 MB
  float*          AHX   = (float*)(ws + 16777216);            // 2048*1088 f32  = 8.9 MB
  float*          Cst   = (float*)(ws + 25690112);            // 32*2048*32 f32 = 8.4 MB

  // re-init recurrent state every call (deterministic across graph replays)
  zero_u32_kernel<<<1048576 / 256, 256, 0, stream>>>((unsigned int*)HXT, 1048576);
  zero_u32_kernel<<<2097152 / 256, 256, 0, stream>>>((unsigned int*)Cst, 2097152);

  adj_softmax_kernel<<<NNODES, 256, 0, stream>>>(E1, E2, Abf);
  pack_x_kernel<<<dim3(NNODES / 256, TSTEPS * BC), 256, 0, stream>>>(x, XallT);

  for (int t = 0; t < TSTEPS; ++t) {
    gemm_wmma_kernel<<<dim3(NCOLS / 64, NNODES / 128), 256, 0, stream>>>(
        Abf, HXT, XallT + (size_t)t * BC * NNODES, AHX);
    gates_lstm_kernel<<<(BATCH * NNODES) / 8, 256, 0, stream>>>(
        AHX, Wx, bx, Wh, bh, Cst, HXT);
  }

  proj_kernel<<<dim3(NNODES / 256, BATCH), 256, 0, stream>>>(HXT, Wp, bp, out);
}
